// NONLocalBlock_23046794510697
// MI455X (gfx1250) — matrix operand order repaired and checked by reading, not compiled, not run on hardware
//
#include <hip/hip_runtime.h>

typedef __attribute__((ext_vector_type(16))) __bf16 v16bf;
typedef __attribute__((ext_vector_type(8)))  float  v8f;
typedef int v4i_ __attribute__((vector_size(16)));
typedef __attribute__((address_space(1))) v4i_ as1_v4i;
typedef __attribute__((address_space(3))) v4i_ as3_v4i;

#define THREADS 256
#define BM 128
#define BN 128
#define BK 32
#define LDSS 40   // padded LDS row stride (bf16 elems) to break bank conflicts

#if defined(__has_builtin)
#  if __has_builtin(__builtin_amdgcn_global_load_async_to_lds_b128) && \
      __has_builtin(__builtin_amdgcn_s_wait_asynccnt)
#    define HAVE_ASYNC_LDS 1
#  endif
#endif
#ifndef HAVE_ASYNC_LDS
#  define HAVE_ASYNC_LDS 0
#endif

union FragB16 { v16bf v; uint4 q[2]; };
union Pack8   { uint4 q; __bf16 h[8]; };

#if HAVE_ASYNC_LDS
// one 16-byte async global->LDS copy (ASYNCcnt-tracked)
__device__ __forceinline__ void async_cp16(const void* gsrc, void* ldst)
{
  __builtin_amdgcn_global_load_async_to_lds_b128(
      (as1_v4i*)gsrc, (as3_v4i*)ldst, 0, 0);
}
#endif

// 8 WMMAs on one staged 128x32 (A) x 32x128 (B as [n][k]) tile pair
__device__ __forceinline__ void compute_tile(const __bf16* __restrict__ As,
                                             const __bf16* __restrict__ Bs,
                                             v8f acc[4][2],
                                             int wm, int wn, int mrow,
                                             int kbA, int kbB)
{
  FragB16 a[4], b[2];
#pragma unroll
  for (int r = 0; r < 4; ++r) {
    int base = (wm + r * 16 + mrow) * LDSS + kbA;
    a[r].q[0] = *(const uint4*)&As[base];        // K = kbA..kbA+7
    a[r].q[1] = *(const uint4*)&As[base + 16];   // K = kbA+16..kbA+23
  }
#pragma unroll
  for (int c = 0; c < 2; ++c) {
    int base = (wn + c * 16 + mrow) * LDSS + kbB;
    b[c].q[0] = *(const uint4*)&Bs[base];        // K = kbB..kbB+7
    b[c].q[1] = *(const uint4*)&Bs[base + 16];   // K = kbB+16..kbB+23
  }
#pragma unroll
  for (int r = 0; r < 4; ++r)
#pragma unroll
    for (int c = 0; c < 2; ++c)
      acc[r][c] = __builtin_amdgcn_wmma_f32_16x16x32_bf16(
          false, a[r].v, false, b[c].v, (short)0, acc[r][c], false, false);
}

// ---------------------------------------------------------------------------
// Tiled bf16 WMMA GEMM: C[M,N] = A[M,K] (bf16 row-major) * B[K,N]
//   B_IS_F32 : global B is f32 (convert to bf16 while staging; implies B_TRANS)
//   B_TRANS  : global B is [K][N] row-major -> transpose-stage into LDS [n][k]
//              (0 means global B is already [N][K] bf16 -> async direct stage)
//   EPI: 0 = f32 C[M,N]                              (scores)
//        1 = bf16 C stored transposed [N,M], +bias   (theta_x/phi_x -> [T,L])
//        2 = bf16 C[M,N], +bias                      (g_x -> [L,T])
//        3 = f32 C[M,N] + per-block LN stat partials (t_x)
//        4 = f32 C[M,N] + bias + residual            (final output)
// ---------------------------------------------------------------------------
template<int B_IS_F32, int B_TRANS, int EPI>
__global__ __launch_bounds__(THREADS)
void gemm_wmma(const __bf16* __restrict__ A,
               const void*   __restrict__ Bvoid,
               void*         __restrict__ Cvoid,
               const float*  __restrict__ bias,
               const float*  __restrict__ residual,
               float*        __restrict__ stats_part,
               int M, int N, int K,
               long strideB, long strideC)
{
  __shared__ __bf16 As[2][BM * LDSS];
  __shared__ __bf16 Bs[2][BN * LDSS];
  __shared__ float  redS[THREADS];
  __shared__ float  redSS[THREADS];

  const int tid = threadIdx.x;
  const int m0  = blockIdx.y * BM;
  const int n0  = blockIdx.x * BN;
  const int z   = blockIdx.z;

  const __bf16* Bb = ((const __bf16*)Bvoid) + (long)z * strideB;
  const float*  Bf = ((const float*) Bvoid) + (long)z * strideB;

  const int wid  = tid >> 5;
  const int lane = tid & 31;
  const int wm   = (wid >> 2) * 64;
  const int wn   = (wid & 3) * 32;
  const int mrow = lane & 15;
  const int kbA  = (lane & 16) ? 8 : 0;
  const int kbB  = (lane & 16) ? 8 : 0;

  // per-thread staging coordinates (two 16B chunks per tile per operand)
  const int arow0 = tid >> 2;                // chunks 0..255
  const int akc0  = (tid & 3) << 3;
  const int arow1 = (tid + THREADS) >> 2;    // chunks 256..511
  const int akc1  = ((tid + THREADS) & 3) << 3;

  v8f acc[4][2];
  const v8f vzero = {0.f,0.f,0.f,0.f,0.f,0.f,0.f,0.f};
#pragma unroll
  for (int r = 0; r < 4; ++r)
#pragma unroll
    for (int c = 0; c < 2; ++c) acc[r][c] = vzero;

  const int nk = K / BK;

#if HAVE_ASYNC_LDS
  if (B_TRANS == 0) {
    // ---- async double-buffered pipeline (ASYNCcnt-tracked copies) ----
    auto stage_async = [&](int k0, int buf) {
      async_cp16(A  + (size_t)(m0 + arow0) * K + k0 + akc0,
                 &As[buf][arow0 * LDSS + akc0]);
      async_cp16(A  + (size_t)(m0 + arow1) * K + k0 + akc1,
                 &As[buf][arow1 * LDSS + akc1]);
      async_cp16(Bb + (size_t)(n0 + arow0) * K + k0 + akc0,
                 &Bs[buf][arow0 * LDSS + akc0]);
      async_cp16(Bb + (size_t)(n0 + arow1) * K + k0 + akc1,
                 &Bs[buf][arow1 * LDSS + akc1]);
    };
    stage_async(0, 0);
    for (int kk = 0; kk < nk; ++kk) {
      if (kk + 1 < nk) {
        stage_async((kk + 1) * BK, (kk + 1) & 1);
        __builtin_amdgcn_s_wait_asynccnt(4);   // tile kk done (in-order)
      } else {
        __builtin_amdgcn_s_wait_asynccnt(0);
      }
      __syncthreads();
      compute_tile(As[kk & 1], Bs[kk & 1], acc, wm, wn, mrow, kbA, kbB);
      __syncthreads();
    }
  } else
#endif
  {
    // ---- synchronous staging (transpose/convert path for f32 B) ----
    for (int kk = 0; kk < nk; ++kk) {
      const int k0 = kk * BK;
      *(uint4*)&As[0][arow0 * LDSS + akc0] =
          *(const uint4*)(A + (size_t)(m0 + arow0) * K + k0 + akc0);
      *(uint4*)&As[0][arow1 * LDSS + akc1] =
          *(const uint4*)(A + (size_t)(m0 + arow1) * K + k0 + akc1);
      if (k0 + BK < K)
        __builtin_prefetch(A + (size_t)(m0 + arow0) * K + k0 + BK, 0, 1);

      if (B_TRANS) {
        if (B_IS_F32) {
#pragma unroll
          for (int i = 0; i < 4; ++i) {
            int c    = tid + i * THREADS;   // 1024 chunks of 4 f32
            int krow = c >> 5;
            int nc   = (c & 31) << 2;
            float4 v = *(const float4*)(Bf + (size_t)(k0 + krow) * N + n0 + nc);
            Bs[0][(nc + 0) * LDSS + krow] = (__bf16)v.x;
            Bs[0][(nc + 1) * LDSS + krow] = (__bf16)v.y;
            Bs[0][(nc + 2) * LDSS + krow] = (__bf16)v.z;
            Bs[0][(nc + 3) * LDSS + krow] = (__bf16)v.w;
          }
        } else {
#pragma unroll
          for (int i = 0; i < 2; ++i) {
            int c    = tid + i * THREADS;   // 512 chunks of 8 bf16
            int krow = c >> 4;
            int nc   = (c & 15) << 3;
            Pack8 p;
            p.q = *(const uint4*)(Bb + (size_t)(k0 + krow) * N + n0 + nc);
#pragma unroll
            for (int e = 0; e < 8; ++e)
              Bs[0][(nc + e) * LDSS + krow] = p.h[e];
          }
        }
      } else {
        *(uint4*)&Bs[0][arow0 * LDSS + akc0] =
            *(const uint4*)(Bb + (size_t)(n0 + arow0) * K + k0 + akc0);
        *(uint4*)&Bs[0][arow1 * LDSS + akc1] =
            *(const uint4*)(Bb + (size_t)(n0 + arow1) * K + k0 + akc1);
      }
      __syncthreads();
      compute_tile(As[0], Bs[0], acc, wm, wn, mrow, kbA, kbB);
      __syncthreads();
    }
  }

  // ---- epilogue ----
  float s = 0.f, ss = 0.f;
#pragma unroll
  for (int r = 0; r < 4; ++r) {
#pragma unroll
    for (int c = 0; c < 2; ++c) {
#pragma unroll
      for (int i = 0; i < 8; ++i) {
        int m = m0 + wm + r * 16 + i + ((lane & 16) ? 8 : 0);
        int n = n0 + wn + c * 16 + (lane & 15);
        float v = acc[r][c][i];
        if (EPI == 0) {
          ((float*)Cvoid + (long)z * strideC)[(size_t)m * N + n] = v;
        } else if (EPI == 1) {
          v += bias[m];
          ((__bf16*)Cvoid + (long)z * strideC)[(size_t)n * M + m] = (__bf16)v;
        } else if (EPI == 2) {
          v += bias[m];
          ((__bf16*)Cvoid + (long)z * strideC)[(size_t)m * N + n] = (__bf16)v;
        } else if (EPI == 3) {
          ((float*)Cvoid)[(size_t)m * N + n] = v;
          s += v; ss += v * v;
        } else { // EPI == 4
          v += bias[m] + residual[(size_t)m * N + n];
          ((float*)Cvoid)[(size_t)m * N + n] = v;
        }
      }
    }
  }
  if (EPI == 3) {
    redS[tid] = s; redSS[tid] = ss;
    __syncthreads();
    for (int o = THREADS / 2; o > 0; o >>= 1) {
      if (tid < o) { redS[tid] += redS[tid + o]; redSS[tid] += redSS[tid + o]; }
      __syncthreads();
    }
    if (tid == 0) {
      int bid = blockIdx.y * gridDim.x + blockIdx.x;
      stats_part[2 * bid]     = redS[0];
      stats_part[2 * bid + 1] = redSS[0];
    }
  }
}

// ---------------------------------------------------------------------------
__global__ void softmax_rows(const float* __restrict__ S,
                             __bf16* __restrict__ P, int N)
{
  __shared__ float red[256];
  const int row = blockIdx.x;
  const float* s = S + (size_t)row * N;
  float m = -3.402823466e38f;
  for (int j = threadIdx.x; j < N; j += 256) m = fmaxf(m, s[j]);
  red[threadIdx.x] = m; __syncthreads();
  for (int o = 128; o > 0; o >>= 1) {
    if (threadIdx.x < o) red[threadIdx.x] = fmaxf(red[threadIdx.x], red[threadIdx.x + o]);
    __syncthreads();
  }
  m = red[0]; __syncthreads();
  float sum = 0.f;
  for (int j = threadIdx.x; j < N; j += 256) sum += __expf(s[j] - m);
  red[threadIdx.x] = sum; __syncthreads();
  for (int o = 128; o > 0; o >>= 1) {
    if (threadIdx.x < o) red[threadIdx.x] += red[threadIdx.x + o];
    __syncthreads();
  }
  float inv = 1.0f / red[0];
  for (int j = threadIdx.x; j < N; j += 256)
    P[(size_t)row * N + j] = (__bf16)(__expf(s[j] - m) * inv);
}

__global__ void finalize_stats(const float* __restrict__ part, int nblocks,
                               float inv_count, float* __restrict__ meanvar)
{
  __shared__ float s0[64], s1[64];
  int t = threadIdx.x;
  float a = 0.f, b = 0.f;
  if (t < nblocks) { a = part[2 * t]; b = part[2 * t + 1]; }
  s0[t] = a; s1[t] = b; __syncthreads();
  for (int o = 32; o > 0; o >>= 1) {
    if (t < o) { s0[t] += s0[t + o]; s1[t] += s1[t + o]; }
    __syncthreads();
  }
  if (t == 0) {
    float mean = s0[0] * inv_count;
    float var  = s1[0] * inv_count - mean * mean;
    meanvar[0] = mean;
    meanvar[1] = rsqrtf(var + 1e-5f);
  }
}

__global__ void ln_relu(const float* __restrict__ tx,
                        const float* __restrict__ gt,
                        const float* __restrict__ bt,
                        const float* __restrict__ meanvar,
                        __bf16* __restrict__ h, int n)
{
  const float mean = meanvar[0], rstd = meanvar[1];
  for (int i = blockIdx.x * blockDim.x + threadIdx.x; i < n;
       i += gridDim.x * blockDim.x) {
    float y = (tx[i] - mean) * rstd * gt[i] + bt[i];
    h[i] = (__bf16)fmaxf(y, 0.f);
  }
}

__global__ void f32_to_bf16(const float* __restrict__ in,
                            __bf16* __restrict__ out, int n)
{
  for (int i = blockIdx.x * blockDim.x + threadIdx.x; i < n;
       i += gridDim.x * blockDim.x)
    out[i] = (__bf16)in[i];
}

// in [R][C] -> out [C][R]
__global__ void transpose_f32(const float* __restrict__ in,
                              float* __restrict__ out, int R, int C)
{
  int i = blockIdx.x * blockDim.x + threadIdx.x;
  if (i < R * C) {
    int r = i / C, c = i % C;
    out[(size_t)c * R + r] = in[i];
  }
}

// ---------------------------------------------------------------------------
extern "C" void kernel_launch(void* const* d_in, const int* in_sizes, int n_in,
                              void* d_out, int out_size, void* d_ws, size_t ws_size,
                              hipStream_t stream)
{
  (void)in_sizes; (void)n_in; (void)out_size; (void)ws_size;
  const float* input1   = (const float*)d_in[0];
  const float* input2   = (const float*)d_in[1];
  const float* theta_w  = (const float*)d_in[2];
  const float* theta_b  = (const float*)d_in[3];
  const float* phi_w    = (const float*)d_in[4];
  const float* phi_b    = (const float*)d_in[5];
  const float* g_w      = (const float*)d_in[6];
  const float* g_b      = (const float*)d_in[7];
  const float* ln_gamma = (const float*)d_in[8];
  const float* ln_beta  = (const float*)d_in[9];
  const float* out_w    = (const float*)d_in[10];
  const float* out_b    = (const float*)d_in[11];
  float* out = (float*)d_out;

  constexpr int BATCH = 16, D = 1024, L = 512, T = 2048, D2 = 1024;

  char* p = (char*)d_ws;
  auto alloc = [&](size_t bytes) -> char* {
    char* r = p; p += (bytes + 255) & ~(size_t)255; return r;
  };
  __bf16* w_theta = (__bf16*)alloc((size_t)L * D * 2);
  __bf16* w_phi   = (__bf16*)alloc((size_t)L * D * 2);
  __bf16* w_g     = (__bf16*)alloc((size_t)L * D * 2);
  __bf16* w_out   = (__bf16*)alloc((size_t)D2 * L * 2);
  __bf16* theta_x = (__bf16*)alloc((size_t)BATCH * T * L * 2);  // [b][T][L]
  __bf16* phi_x   = (__bf16*)alloc((size_t)BATCH * T * L * 2);  // [b][T][L] (phi^T)
  __bf16* g_x     = (__bf16*)alloc((size_t)BATCH * L * T * 2);  // [b][L][T] (g^T of btl)
  float*  gt      = (float*) alloc((size_t)T * L * 4);          // gamma^T [T][L]
  float*  bt      = (float*) alloc((size_t)T * L * 4);          // beta^T  [T][L]
  float*  scores  = (float*) alloc((size_t)T * T * 4);          // per-batch reuse
  __bf16* pmat    = (__bf16*)alloc((size_t)T * T * 2);          // per-batch reuse
  float*  t_x     = (float*) alloc((size_t)T * L * 4);          // per-batch [T][L]
  __bf16* h       = (__bf16*)alloc((size_t)T * L * 2);          // per-batch [T][L]
  float*  stats   = (float*) alloc(64 * 2 * 4);
  float*  meanvar = (float*) alloc(2 * 4);

  // weight conversion + gamma/beta transpose
  f32_to_bf16<<<(L * D) / 256, 256, 0, stream>>>(theta_w, w_theta, L * D);
  f32_to_bf16<<<(L * D) / 256, 256, 0, stream>>>(phi_w,   w_phi,   L * D);
  f32_to_bf16<<<(L * D) / 256, 256, 0, stream>>>(g_w,     w_g,     L * D);
  f32_to_bf16<<<(D2 * L) / 256, 256, 0, stream>>>(out_w,  w_out,   D2 * L);
  transpose_f32<<<(L * T) / 256, 256, 0, stream>>>(ln_gamma, gt, L, T);
  transpose_f32<<<(L * T) / 256, 256, 0, stream>>>(ln_beta,  bt, L, T);

  dim3 blk(THREADS);
  {
    dim3 grd(T / BN, L / BM, BATCH);
    // theta_x[b] = (theta_w @ input2[b])^T -> [T,L] bf16
    gemm_wmma<1, 1, 1><<<grd, blk, 0, stream>>>(
        w_theta, input2, theta_x, theta_b, nullptr, nullptr,
        L, T, D, (long)D * T, (long)L * T);
    // phi_x[b] = (phi_w @ input1[b])^T -> [T,L] bf16 ([s][l], direct-B for scores)
    gemm_wmma<1, 1, 1><<<grd, blk, 0, stream>>>(
        w_phi, input1, phi_x, phi_b, nullptr, nullptr,
        L, T, D, (long)D * T, (long)L * T);
    // g_x[b] = g_w @ input1[b] -> [L,T] bf16 ([l][s], direct-B for apply)
    gemm_wmma<1, 1, 2><<<grd, blk, 0, stream>>>(
        w_g, input1, g_x, g_b, nullptr, nullptr,
        L, T, D, (long)D * T, (long)L * T);
  }

  for (int b = 0; b < BATCH; ++b) {
    const __bf16* th = theta_x + (size_t)b * T * L;
    const __bf16* ph = phi_x   + (size_t)b * T * L;
    const __bf16* gx = g_x     + (size_t)b * L * T;

    // scores[T,T] = theta_x[T,L] @ phi (B = phi^T [T,L] direct)  (f32)
    {
      dim3 grd(T / BN, T / BM, 1);
      gemm_wmma<0, 0, 0><<<grd, blk, 0, stream>>>(
          th, ph, scores, nullptr, nullptr, nullptr,
          T, T, L, 0L, 0L);
    }
    softmax_rows<<<T, 256, 0, stream>>>(scores, pmat, T);

    // t_x[T,L] = p[T,T] @ g (B = g^T [L,T] direct)  (f32) + LN stat partials
    {
      dim3 grd(L / BN, T / BM, 1);  // 4 x 16 = 64 blocks
      gemm_wmma<0, 0, 3><<<grd, blk, 0, stream>>>(
          pmat, gx, t_x, nullptr, nullptr, stats,
          T, L, T, 0L, 0L);
    }
    finalize_stats<<<1, 64, 0, stream>>>(stats, 64, 1.0f / (float)(T * L), meanvar);

    ln_relu<<<1024, 256, 0, stream>>>(t_x, gt, bt, meanvar, h, T * L);

    // out[b] = out_w[D2,L] @ h (B = h [T,L] direct) + out_b + input2[b]
    {
      dim3 grd(T / BN, D2 / BM, 1);
      gemm_wmma<0, 0, 4><<<grd, blk, 0, stream>>>(
          w_out, h, out + (size_t)b * D2 * T, out_b,
          input2 + (size_t)b * D2 * T, nullptr,
          D2, T, L, 0L, 0L);
    }
  }
}
